// ProteinFeatures_41188736368931
// MI455X (gfx1250) — matrix operand order repaired and checked by reading, not compiled, not run on hardware
//
#include <hip/hip_runtime.h>
#include <hip/hip_bf16.h>

// ---------------- problem constants (match setup_inputs) ----------------
#define BATCH    4
#define LSEQ     2048
#define TOPK     48
#define NUM_RBF  16
#define NUM_POS  16
#define HIDDEN   128
#define KDIM     416          // 25*16 + 16
#define NEDGES   (BATCH * LSEQ * TOPK)    // 393216
#define EDGES_PER_BLK 64
#define LDS_STRIDE 424        // padded row stride (bf16 elems), keeps 16B align, avoids bank conflicts

typedef __attribute__((ext_vector_type(16))) __bf16 v16bf;
typedef __attribute__((ext_vector_type(8)))  float  v8f;

__device__ __forceinline__ unsigned short f32_to_bf16_rne(float f) {
    unsigned int u = __float_as_uint(f);
    u += 0x7FFFu + ((u >> 16) & 1u);     // round to nearest even
    return (unsigned short)(u >> 16);
}

// ---------------- kernel 0: W_edge (416x128 f32) -> W_edgeT (128x416 bf16) ----------------
__global__ __launch_bounds__(256) void k_wedgeT(const float* __restrict__ W_edge,
                                                unsigned short* __restrict__ WT) {
    int idx = blockIdx.x * 256 + threadIdx.x;
    if (idx >= HIDDEN * KDIM) return;
    int n = idx / KDIM, k = idx % KDIM;
    WT[n * KDIM + k] = f32_to_bf16_rne(W_edge[k * HIDDEN + n]);
}

// ---------------- kernel 1: atoms_5 = [N,CA,C,O,CB], plus contiguous CA ----------------
__global__ __launch_bounds__(256) void k_atoms(const float* __restrict__ X,
                                               float* __restrict__ atoms5,
                                               float* __restrict__ CAout) {
    int idx = blockIdx.x * 256 + threadIdx.x;      // B*L residues
    if (idx >= BATCH * LSEQ) return;
    const float* p = X + (long long)idx * 12;      // 4 atoms x 3
    float Nx = p[0], Ny = p[1], Nz = p[2];
    float Ax = p[3], Ay = p[4], Az = p[5];         // CA
    float Cx = p[6], Cy = p[7], Cz = p[8];
    float Ox = p[9], Oy = p[10], Oz = p[11];
    float bx = Ax - Nx, by = Ay - Ny, bz = Az - Nz;
    float cx = Cx - Ax, cy = Cy - Ay, cz = Cz - Az;
    float ax = by * cz - bz * cy;
    float ay = bz * cx - bx * cz;
    float az = bx * cy - by * cx;
    const float kA = -0.58273431f, kB = 0.56802827f, kC = -0.54067466f;
    float CBx = kA * ax + kB * bx + kC * cx + Ax;
    float CBy = kA * ay + kB * by + kC * cy + Ay;
    float CBz = kA * az + kB * bz + kC * cz + Az;
    float* o = atoms5 + (long long)idx * 15;
    o[0]=Nx; o[1]=Ny; o[2]=Nz;  o[3]=Ax; o[4]=Ay; o[5]=Az;
    o[6]=Cx; o[7]=Cy; o[8]=Cz;  o[9]=Ox; o[10]=Oy; o[11]=Oz;
    o[12]=CBx; o[13]=CBy; o[14]=CBz;
    CAout[idx*3+0] = Ax; CAout[idx*3+1] = Ay; CAout[idx*3+2] = Az;
}

// ---------------- kernel 2: top-48 nearest neighbors; one wave32 per residue ----------------
__global__ __launch_bounds__(256) void k_topk(const float* __restrict__ CA,
                                              const float* __restrict__ mask,
                                              int* __restrict__ Eidx) {
    int lane = threadIdx.x & 31;
    int res  = blockIdx.x * 8 + (threadIdx.x >> 5);   // 8 waves per block
    int b = res >> 11;                                 // L = 2048
    int i = res & (LSEQ - 1);
    const float* CAb = CA + (long long)b * LSEQ * 3;
    const float* mb  = mask + (long long)b * LSEQ;
    float xi = CAb[i*3+0], yi = CAb[i*3+1], zi = CAb[i*3+2];

    float dist[64];                                    // 64*32 = 2048 candidates
#pragma unroll
    for (int k = 0; k < 64; k++) {
        int j = k * 32 + lane;
        float dx = xi - CAb[j*3+0];
        float dy = yi - CAb[j*3+1];
        float dz = zi - CAb[j*3+2];
        dist[k] = dx*dx + dy*dy + dz*dz + (1.0f - mb[j]) * 1.0e6f;
    }

    int* out = Eidx + (long long)res * TOPK;
    for (int r = 0; r < TOPK; r++) {
        float vmin = 3.0e38f; int imin = 0x7fffffff;
#pragma unroll
        for (int k = 0; k < 64; k++) {
            int j = k * 32 + lane;
            bool better = dist[k] < vmin;              // strict: keeps lowest index on tie
            vmin = better ? dist[k] : vmin;
            imin = better ? j : imin;
        }
#pragma unroll
        for (int m = 16; m >= 1; m >>= 1) {
            float ov = __shfl_xor(vmin, m, 32);
            int   oi = __shfl_xor(imin, m, 32);
            if (ov < vmin || (ov == vmin && oi < imin)) { vmin = ov; imin = oi; }
        }
        if (lane == 0) out[r] = imin;
#pragma unroll
        for (int k = 0; k < 64; k++)
            if (k * 32 + lane == imin) dist[k] = 3.0e38f;
    }
}

// ---------------- kernel 3: edge features (bf16 LDS) + WMMA GEMM + LayerNorm ----------------
__global__ __launch_bounds__(128) void k_edges(const float* __restrict__ atoms5,
                                               const int* __restrict__ Ridx,
                                               const int* __restrict__ chain,
                                               const float* __restrict__ Wpos,
                                               const float* __restrict__ bpos,
                                               const unsigned short* __restrict__ WT, // 128x416 bf16
                                               const float* __restrict__ gamma,
                                               const float* __restrict__ beta,
                                               const int* __restrict__ Eidx,
                                               float* __restrict__ Eout) {
    __shared__ unsigned short lds[EDGES_PER_BLK * LDS_STRIDE];  // 54,272 B
    const int tid = threadIdx.x;
    const int eBase = blockIdx.x * EDGES_PER_BLK;

    // ---- RBF features: 64 edges * 25 atom pairs = 1600 tasks ----
    for (int s = 0; s < 13; s++) {
        int task = tid + s * 128;
        if (task < EDGES_PER_BLK * 25) {
            int le = task / 25, p = task % 25;
            int ac = p / 5, an = p % 5;               // center atom, neighbor atom
            int e = eBase + le;
            int b   = e / (LSEQ * TOPK);
            int rem = e % (LSEQ * TOPK);
            int i   = rem / TOPK;
            int j   = Eidx[e];
            const float* pa = atoms5 + ((long long)(b * LSEQ + i) * 5 + ac) * 3;
            const float* pb = atoms5 + ((long long)(b * LSEQ + j) * 5 + an) * 3;
            float dx = pa[0] - pb[0], dy = pa[1] - pb[1], dz = pa[2] - pb[2];
            float D = sqrtf(dx*dx + dy*dy + dz*dz + 1e-6f);
            unsigned int pk[8];
#pragma unroll
            for (int r = 0; r < 8; r++) {
                float mu0 = 2.0f + (float)(2*r)   * (20.0f / 15.0f);   // linspace(2,22,16)
                float mu1 = 2.0f + (float)(2*r+1) * (20.0f / 15.0f);
                float t0 = (D - mu0) * 0.8f;                           // 1/sigma, sigma=1.25
                float t1 = (D - mu1) * 0.8f;
                unsigned short lo = f32_to_bf16_rne(__expf(-t0*t0));
                unsigned short hi = f32_to_bf16_rne(__expf(-t1*t1));
                pk[r] = (unsigned int)lo | ((unsigned int)hi << 16);
            }
            // 16B-aligned store (two uint4) of 16 bf16 at column 16 + p*16
            uint4* dst = (uint4*)&lds[le * LDS_STRIDE + NUM_POS + p * 16];
            dst[0] = make_uint4(pk[0], pk[1], pk[2], pk[3]);
            dst[1] = make_uint4(pk[4], pk[5], pk[6], pk[7]);
        }
    }

    // ---- positional features: 64 edges * 16 = 1024 tasks ----
    for (int s = 0; s < 8; s++) {
        int task = tid + s * 128;
        int le = task / NUM_POS, pp = task % NUM_POS;
        int e = eBase + le;
        int b   = e / (LSEQ * TOPK);
        int rem = e % (LSEQ * TOPK);
        int i   = rem / TOPK;
        int j   = Eidx[e];
        int off = Ridx[b * LSEQ + i] - Ridx[b * LSEQ + j];
        int same = (chain[b * LSEQ + i] == chain[b * LSEQ + j]);
        int dd = off + 32;
        dd = dd < 0 ? 0 : (dd > 64 ? 64 : dd);
        dd = same ? dd : 65;
        lds[le * LDS_STRIDE + pp] = f32_to_bf16_rne(Wpos[dd * NUM_POS + pp] + bpos[pp]);
    }
    __syncthreads();

    // ---- GEMM: each wave computes a 16x128 tile via v_wmma_f32_16x16x32_bf16 ----
    const int wv = tid >> 5, lane = tid & 31;
    const int l16 = lane & 15;
    const int hi  = lane >> 4;              // 0 / 1
    const int kbA = hi * 8;                 // A-frag K sub-offset per ISA A layout
    const int kbB = hi * 8;                 // B-frag K sub-offset, same K order as A
    const unsigned short* arow = &lds[(wv * 16 + l16) * LDS_STRIDE];

    float g[8], bt[8];
#pragma unroll
    for (int t = 0; t < 8; t++) { int n = t * 16 + l16; g[t] = gamma[n]; bt[t] = beta[n]; }

    union U { uint4 u[2]; v16bf v; };
    v8f zero = {0.f,0.f,0.f,0.f,0.f,0.f,0.f,0.f};
    v8f acc[8];
#pragma unroll
    for (int t = 0; t < 8; t++) acc[t] = zero;

    for (int kk = 0; kk < 13; kk++) {
        int K0 = kk * 32;
        U a;
        a.u[0] = *(const uint4*)(arow + K0 + kbA);        // K = K0+kb .. +7
        a.u[1] = *(const uint4*)(arow + K0 + 16 + kbA);   // K = K0+16+kb .. +7
#pragma unroll
        for (int t = 0; t < 8; t++) {
            const unsigned short* brow = WT + (t * 16 + l16) * KDIM + K0 + kbB;
            U bm;
            bm.u[0] = *(const uint4*)(brow);              // K = K0+kbB .. +7
            bm.u[1] = *(const uint4*)(brow + 16);         // K = K0+16+kbB .. +7
            acc[t] = __builtin_amdgcn_wmma_f32_16x16x32_bf16(
                         false, a.v, false, bm.v, (short)0, acc[t], false, false);
        }
    }

    // ---- LayerNorm epilogue; C/D layout: VGPR r -> M = r + 8*hi, N = 16t + l16 ----
#pragma unroll
    for (int r = 0; r < 8; r++) {
        float s = 0.f, s2 = 0.f;
#pragma unroll
        for (int t = 0; t < 8; t++) { float x = acc[t][r]; s += x; s2 += x * x; }
#pragma unroll
        for (int m = 8; m >= 1; m >>= 1) {            // reduce within 16-lane half (same M)
            s  += __shfl_xor(s,  m, 32);
            s2 += __shfl_xor(s2, m, 32);
        }
        float mean = s * (1.0f / 128.0f);
        float var  = s2 * (1.0f / 128.0f) - mean * mean;
        float inv  = rsqrtf(var + 1e-5f);
        int eg = eBase + wv * 16 + r + hi * 8;        // global edge index
        float* orow = Eout + (long long)eg * HIDDEN + l16;
#pragma unroll
        for (int t = 0; t < 8; t++)
            orow[t * 16] = (acc[t][r] - mean) * inv * g[t] + bt[t];
    }
}

// ---------------- launch ----------------
extern "C" void kernel_launch(void* const* d_in, const int* in_sizes, int n_in,
                              void* d_out, int out_size, void* d_ws, size_t ws_size,
                              hipStream_t stream) {
    const float* X      = (const float*)d_in[0];
    const float* rmask  = (const float*)d_in[1];
    const int*   Ridx   = (const int*)d_in[2];
    const int*   chain  = (const int*)d_in[3];
    const float* Wpos   = (const float*)d_in[4];
    const float* bpos   = (const float*)d_in[5];
    const float* Wedge  = (const float*)d_in[6];
    const float* gamma  = (const float*)d_in[7];
    const float* beta   = (const float*)d_in[8];

    // d_out = [E_idx int32 (B*L*48)] ++ [E f32 (B*L*48*128)]
    int*   Eidx = (int*)d_out;
    float* E    = (float*)d_out + NEDGES;

    // workspace layout (16B-aligned sections)
    char* ws = (char*)d_ws;
    float* atoms5 = (float*)(ws);                                  // B*L*15 f32 = 491,520 B
    float* CAb    = (float*)(ws + 491520);                         // B*L*3  f32 =  98,304 B
    unsigned short* WT = (unsigned short*)(ws + 491520 + 98304);   // 128*416 bf16 = 106,496 B

    k_wedgeT<<<(HIDDEN * KDIM + 255) / 256, 256, 0, stream>>>(Wedge, WT);
    k_atoms <<<(BATCH * LSEQ + 255) / 256, 256, 0, stream>>>(X, atoms5, CAb);
    k_topk  <<<(BATCH * LSEQ) / 8, 256, 0, stream>>>(CAb, rmask, Eidx);
    k_edges <<<NEDGES / EDGES_PER_BLK, 128, 0, stream>>>(atoms5, Ridx, chain, Wpos, bpos,
                                                         WT, gamma, beta, Eidx, E);
}
